// Model_3307124818158
// MI455X (gfx1250) — compile-verified, matrix operand order checked by reading, not run on hardware
//
#include <hip/hip_runtime.h>
#include <hip/hip_bf16.h>

typedef _Float16 f16;
typedef __attribute__((ext_vector_type(16))) _Float16 v16h;
typedef __attribute__((ext_vector_type(8)))  _Float16 v8h;
typedef __attribute__((ext_vector_type(8)))  float    v8f;

#define B_  256
#define T_  2048
#define V_  128
#define H_  64
#define S_  (V_ + H_)     // 192
#define TILE_M 16
#define NWAVES 8
#define NTHREADS (NWAVES * 32)   // 256

union AFrag { v16h v; v8h h[2]; };
union Acc   { v8f  v; float f[8]; };

__device__ __forceinline__ float tanh_fast(float x) {
    x = fminf(fmaxf(x, -15.0f), 15.0f);
    float e = __expf(2.0f * x);
    return (e - 1.0f) / (e + 1.0f);
}
// Branchless unified activation: result = (a*e + b) / (e + 1), e = exp(s*x)
//   tanh    : s= 2, a=1, b=-1
//   sigmoid : s=-1, a=0, b= 1
__device__ __forceinline__ float act_fast(float x, float s, float a, float b) {
    x = fminf(fmaxf(x, -30.0f), 30.0f);
    float e = __expf(s * x);
    return (a * e + b) / (e + 1.0f);
}

// Async copy of 16 bytes global->LDS (GLOBAL_LOAD_ASYNC_TO_LDS_B128, GVS mode).
__device__ __forceinline__ void async_copy_b128(uint32_t lds_off, uint64_t gbase, uint32_t byte_off) {
    asm volatile("global_load_async_to_lds_b128 %0, %1, %2"
                 :: "v"(lds_off), "v"(byte_off), "s"(gbase)
                 : "memory");
}
__device__ __forceinline__ void wait_async0() {
    asm volatile("s_wait_asynccnt 0" ::: "memory");
}

// Stage x[row0..row0+15, t, :] (16 x 128 f32 = 512 b128 chunks) into LDS asynchronously.
__device__ __forceinline__ void stage_x_async(const float* __restrict__ x, int row0, int t,
                                              const char* smemBase, float* dstLds, int tid) {
    uint32_t dst0 = (uint32_t)((uintptr_t)dstLds - (uintptr_t)smemBase);
    uint64_t gbase = (uint64_t)(uintptr_t)(x + ((size_t)row0 * T_ + t) * V_);
    #pragma unroll
    for (int j = 0; j < 512 / NTHREADS; ++j) {          // 2 chunks per thread
        int i = tid + j * NTHREADS;
        int r = i >> 5, c = i & 31;                     // row, 16B-chunk within row
        uint32_t ldsoff = dst0 + (uint32_t)(r * V_ + c * 4) * 4u;
        uint32_t goff   = (uint32_t)((size_t)r * T_ * V_ + (size_t)c * 4) * 4u;
        async_copy_b128(ldsoff, gbase, goff);
    }
}

__global__ __launch_bounds__(NTHREADS)
void lstm_scan_wmma(const float* __restrict__ x,
                    const float* __restrict__ W_f,  const float* __restrict__ b_f,
                    const float* __restrict__ W_if, const float* __restrict__ b_if,
                    const float* __restrict__ W_ic, const float* __restrict__ b_ic,
                    const float* __restrict__ W_o,  const float* __restrict__ b_o,
                    const float* __restrict__ W_out,const float* __restrict__ b_out,
                    float* __restrict__ out) {
    extern __shared__ char smem[];
    // LDS carve-up (all 16B aligned)
    f16*   Wt   = (f16*)smem;                    // [4][64][192]  gate weights, col-major (B-frag friendly)
    f16*   WoT  = Wt + 4 * H_ * S_;              // [128][64]     W_out transposed
    float* bgS  = (float*)(WoT + V_ * H_);       // [4][64]
    float* boS  = bgS + 4 * H_;                  // [128]
    f16*   sM   = (f16*)(boS + V_);              // [16][192]     s = concat(x_t, h) in f16
    float* gact = (float*)(sM + TILE_M * S_);    // [4][16][64]   activated gates
    float* cst  = gact + 4 * TILE_M * H_;        // [16][64]      cell state
    float* lgt  = cst + TILE_M * H_;             // [16][128]     logits / exp scratch
    float* xstg = lgt + TILE_M * V_;             // [2][16][128]  async x double-buffer (f32)

    const int tid  = threadIdx.x;
    const int wave = tid >> 5;
    const int lane = tid & 31;
    const int lr   = lane & 15;       // row (A) / col (B/D) within tile
    const int hi   = lane >> 4;       // half selector for K chunks / M offset
    const int gate = wave >> 1;       // 0..3: which gate this wave computes
    const int nh   = wave & 1;        // which N-half (2 tiles of 16) of the gate
    const int row0 = blockIdx.x * TILE_M;

    // wave-uniform activation coefficients (gate 2 = candidate -> tanh, else sigmoid)
    const float actS = (gate == 2) ?  2.0f : -1.0f;
    const float actA = (gate == 2) ?  1.0f :  0.0f;
    const float actB = (gate == 2) ? -1.0f :  1.0f;

    // ---- one-time weight staging: f32 -> f16, transposed for contiguous B-fragments ----
    const float* Wg[4]  = { W_f, W_if, W_ic, W_o };
    const float* bgp[4] = { b_f, b_if, b_ic, b_o };
    for (int g = 0; g < 4; ++g)
        for (int i = tid; i < H_ * S_; i += NTHREADS) {
            int col = i / S_, k = i % S_;
            Wt[g * H_ * S_ + i] = (f16)Wg[g][k * H_ + col];
        }
    for (int i = tid; i < V_ * H_; i += NTHREADS) {
        int n = i / H_, k = i % H_;
        WoT[i] = (f16)W_out[k * V_ + n];
    }
    for (int g = 0; g < 4; ++g)
        for (int i = tid; i < H_; i += NTHREADS) bgS[g * H_ + i] = bgp[g][i];
    for (int i = tid; i < V_; i += NTHREADS) boS[i] = b_out[i];
    for (int i = tid; i < TILE_M * H_; i += NTHREADS) {
        cst[i] = 0.0f;
        sM[(i >> 6) * S_ + V_ + (i & 63)] = (f16)0.0f;   // h0 = 0
    }

    // kick off async copy of the first x tile while we finish staging weights
    stage_x_async(x, row0, 0, smem, xstg, tid);
    __syncthreads();

    // ---- hoist ALL loop-invariant operands into registers ----
    // gate weights: 6 K-steps x 2 N-tiles = 12 frags (96 VGPRs) per wave
    AFrag gateB[6][2];
    #pragma unroll
    for (int kk = 0; kk < 6; ++kk)
        #pragma unroll
        for (int n = 0; n < 2; ++n) {
            const f16* brow = Wt + gate * H_ * S_ + ((nh * 2 + n) * 16 + lr) * S_ + kk * 32 + hi * 8;
            gateB[kk][n].h[0] = *(const v8h*)(brow);
            gateB[kk][n].h[1] = *(const v8h*)(brow + 16);
        }
    // W_out: wave handles N-tile `wave` of 8; 2 K-steps = 2 frags (16 VGPRs)
    AFrag woutB[2];
    #pragma unroll
    for (int kk = 0; kk < 2; ++kk) {
        const f16* brow = WoT + (wave * 16 + lr) * H_ + kk * 32 + hi * 8;
        woutB[kk].h[0] = *(const v8h*)(brow);
        woutB[kk].h[1] = *(const v8h*)(brow + 16);
    }
    // per-lane biases (loop-invariant)
    float gbias[2];
    #pragma unroll
    for (int n = 0; n < 2; ++n) gbias[n] = bgS[gate * H_ + (nh * 2 + n) * 16 + lr];
    const float obias = boS[wave * 16 + lr];

    // ---- sequential scan over time ----
    #pragma unroll 1
    for (int t = 0; t < T_; ++t) {
        wait_async0();                       // x tile for step t is in LDS
        __syncthreads();
        float* xs = xstg + (t & 1) * TILE_M * V_;
        if (t + 1 < T_)                      // overlap next tile's HBM fetch with this step
            stage_x_async(x, row0, t + 1, smem, xstg + ((t + 1) & 1) * TILE_M * V_, tid);

        // convert staged f32 -> f16 into sM[:, 0:128)
        #pragma unroll
        for (int j = 0; j < 512 / NTHREADS; ++j) {
            int i = tid + j * NTHREADS;                  // 512 float4 chunks
            const float4 v = *(const float4*)(xs + i * 4);
            int r = i >> 5, c4 = i & 31;
            f16* dst = sM + r * S_ + c4 * 4;
            dst[0] = (f16)v.x; dst[1] = (f16)v.y; dst[2] = (f16)v.z; dst[3] = (f16)v.w;
        }
        __syncthreads();

        // ---- gate GEMM: wave computes its gate / N-half: [16,192]x[192,32] ----
        // 2 N-tiles x 6 K-steps; B-fragments resident in registers, only A from LDS
        {
            Acc acc[2];
            #pragma unroll
            for (int n = 0; n < 2; ++n)
                #pragma unroll
                for (int j = 0; j < 8; ++j) acc[n].f[j] = 0.0f;

            #pragma unroll
            for (int kk = 0; kk < 6; ++kk) {
                AFrag a;
                const f16* arow = sM + lr * S_ + kk * 32 + hi * 8;
                a.h[0] = *(const v8h*)(arow);
                a.h[1] = *(const v8h*)(arow + 16);
                #pragma unroll
                for (int n = 0; n < 2; ++n)
                    acc[n].v = __builtin_amdgcn_wmma_f32_16x16x32_f16(
                        false, a.v, false, gateB[kk][n].v, (short)0, acc[n].v, false, false);
            }
            // bias + branchless activation, scatter to LDS
            // (D layout: M = r + 8*hi, N-col = tile*16 + lr)
            #pragma unroll
            for (int n = 0; n < 2; ++n)
                #pragma unroll
                for (int r = 0; r < 8; ++r) {
                    int m = r + hi * 8, col = (nh * 2 + n) * 16 + lr;
                    float vv = act_fast(acc[n].f[r] + gbias[n], actS, actA, actB);
                    gact[(gate * TILE_M + m) * H_ + col] = vv;
                }
        }
        __syncthreads();

        // ---- elementwise cell update: c = c*f + add*af ; h = tanh(c)*o ----
        #pragma unroll
        for (int j = 0; j < (TILE_M * H_) / NTHREADS; ++j) {
            int i = tid + j * NTHREADS;
            int r = i >> 6, col = i & 63;
            float fg  = gact[(0 * TILE_M + r) * H_ + col];
            float afg = gact[(1 * TILE_M + r) * H_ + col];
            float ag  = gact[(2 * TILE_M + r) * H_ + col];
            float og  = gact[(3 * TILE_M + r) * H_ + col];
            float c   = cst[i] * fg + ag * afg;
            cst[i] = c;
            sM[r * S_ + V_ + col] = (f16)(tanh_fast(c) * og);  // h (f16) for out-GEMM + next step
        }
        __syncthreads();

        // ---- output GEMM: logits[16,128] = h[16,64] @ W_out[64,128] + b_out ----
        // wave handles N-tile `wave`; K=64 -> 2 K-steps; B-frags in registers
        {
            Acc oacc;
            #pragma unroll
            for (int j = 0; j < 8; ++j) oacc.f[j] = 0.0f;

            #pragma unroll
            for (int kk = 0; kk < 2; ++kk) {
                AFrag a;
                const f16* arow = sM + lr * S_ + V_ + kk * 32 + hi * 8;
                a.h[0] = *(const v8h*)(arow);
                a.h[1] = *(const v8h*)(arow + 16);
                oacc.v = __builtin_amdgcn_wmma_f32_16x16x32_f16(
                    false, a.v, false, woutB[kk].v, (short)0, oacc.v, false, false);
            }
            #pragma unroll
            for (int r = 0; r < 8; ++r) {
                int m = r + hi * 8, col = wave * 16 + lr;
                lgt[m * V_ + col] = oacc.f[r] + obias;
            }
        }
        __syncthreads();

        // ---- softmax over V=128 per row, 16 threads/row, wave32 shfl reductions ----
        {
            int r = tid >> 4, sub = tid & 15;
            float m = -1e30f;
            for (int c = sub; c < V_; c += 16) m = fmaxf(m, lgt[r * V_ + c]);
            #pragma unroll
            for (int off = 8; off; off >>= 1) m = fmaxf(m, __shfl_xor(m, off, 16));
            float ssum = 0.0f;
            for (int c = sub; c < V_; c += 16) {
                float e = __expf(lgt[r * V_ + c] - m);
                lgt[r * V_ + c] = e;
                ssum += e;
            }
            #pragma unroll
            for (int off = 8; off; off >>= 1) ssum += __shfl_xor(ssum, off, 16);
            float inv = 1.0f / ssum;
            float* orow = out + (((size_t)(row0 + r)) * T_ + t) * V_;
            for (int c = sub; c < V_; c += 16) orow[c] = lgt[r * V_ + c] * inv;
        }
        __syncthreads();
    }
}

extern "C" void kernel_launch(void* const* d_in, const int* in_sizes, int n_in,
                              void* d_out, int out_size, void* d_ws, size_t ws_size,
                              hipStream_t stream) {
    (void)in_sizes; (void)n_in; (void)out_size; (void)d_ws; (void)ws_size;
    const float* xp    = (const float*)d_in[0];
    const float* W_f   = (const float*)d_in[1];
    const float* b_f   = (const float*)d_in[2];
    const float* W_if  = (const float*)d_in[3];
    const float* b_if  = (const float*)d_in[4];
    const float* W_ic  = (const float*)d_in[5];
    const float* b_ic  = (const float*)d_in[6];
    const float* W_o   = (const float*)d_in[7];
    const float* b_o   = (const float*)d_in[8];
    const float* W_out = (const float*)d_in[9];
    const float* b_out = (const float*)d_in[10];
    float* outp = (float*)d_out;

    size_t smem = (size_t)(4 * H_ * S_) * sizeof(f16)          // gate weights
                + (size_t)(V_ * H_) * sizeof(f16)              // W_out^T
                + (size_t)(4 * H_ + V_) * sizeof(float)        // biases
                + (size_t)(TILE_M * S_) * sizeof(f16)          // s buffer
                + (size_t)(4 * TILE_M * H_) * sizeof(float)    // gate activations
                + (size_t)(TILE_M * H_) * sizeof(float)        // cell state
                + (size_t)(TILE_M * V_) * sizeof(float)        // logits
                + (size_t)(2 * TILE_M * V_) * sizeof(float);   // async x double-buffer

    dim3 grid(B_ / TILE_M, 1, 1);
    dim3 block(NTHREADS, 1, 1);
    lstm_scan_wmma<<<grid, block, smem, stream>>>(
        xp, W_f, b_f, W_if, b_if, W_ic, b_ic, W_o, b_o, W_out, b_out, outp);
}
